// PFNet_61667140436153
// MI455X (gfx1250) — matrix operand order repaired and checked by reading, not compiled, not run on hardware
//
#include <hip/hip_runtime.h>
#include <hip/hip_bf16.h>

// ---------------- constants ----------------
#define BB   2
#define NNP  4096
#define BNP  (BB*NNP)
#define FFP  16
#define HIDP 256
#define ENCD 27
#define H2P  512
#define H2CP 520
#define LDP  544   // row stride for all big activation buffers (bf16)

typedef __attribute__((ext_vector_type(16))) __bf16 bf16x16;
typedef __attribute__((ext_vector_type(8)))  float  f32x8;
typedef __attribute__((ext_vector_type(8)))  unsigned int u32x8;

// ---------------- helpers ----------------
__device__ __forceinline__ unsigned short f2bf(float f) {
  unsigned int u = __builtin_bit_cast(unsigned int, f);
  u += 0x7FFFu + ((u >> 16) & 1u);            // round-to-nearest-even
  return (unsigned short)(u >> 16);
}
__device__ __forceinline__ float bf2f(unsigned short h) {
  unsigned int u = ((unsigned int)h) << 16;
  return __builtin_bit_cast(float, u);
}
__device__ __forceinline__ float selu_f(float x) {
  const float sc = 1.0507009873554805f, al = 1.6732632423543772f;
  return x > 0.f ? sc * x : sc * al * (__expf(x) - 1.f);
}
// build a 16-element bf16 fragment from two 16B-aligned LDS chunks
__device__ __forceinline__ bf16x16 frag2(const unsigned short* p0,
                                         const unsigned short* p1) {
  uint4 a = *(const uint4*)p0;
  uint4 b = *(const uint4*)p1;
  u32x8 v;
  v[0] = a.x; v[1] = a.y; v[2] = a.z; v[3] = a.w;
  v[4] = b.x; v[5] = b.y; v[6] = b.z; v[7] = b.w;
  return __builtin_bit_cast(bf16x16, v);
}

// ---------------- tiny elementwise kernels ----------------
__global__ void cvt_f32_bf16_kernel(const float* __restrict__ s,
                                    unsigned short* __restrict__ d, long n) {
  long i = (long)blockIdx.x * blockDim.x + threadIdx.x;
  if (i < n) d[i] = f2bf(s[i]);
}

__global__ void zero_f32_kernel(float* __restrict__ p, long n) {
  long i = (long)blockIdx.x * blockDim.x + threadIdx.x;
  if (i < n) p[i] = 0.f;
}

__global__ void encode_kernel(const float* __restrict__ X,
                              unsigned short* __restrict__ enc,
                              float* __restrict__ msk, int bn) {
  int i = blockIdx.x * blockDim.x + threadIdx.x;
  if (i >= bn) return;
  const float* x = X + (long)i * FFP;
  float id = x[0];
  msk[i] = (id != 0.f) ? 1.f : 0.f;
  int idi = (int)id;
  unsigned short* e = enc + (long)i * 32;
  const float s1 = 1.0507009873554805f;   // selu(1)
#pragma unroll
  for (int j = 0; j < 12; ++j) e[j] = f2bf(j == idi ? s1 : 0.f);
#pragma unroll
  for (int j = 0; j < 15; ++j) e[12 + j] = f2bf(selu_f(x[1 + j]));
#pragma unroll
  for (int j = 27; j < 32; ++j) e[j] = 0;
}

__global__ void rowsq_kernel(const unsigned short* __restrict__ c, long ld,
                             float* __restrict__ na, int rows, int D) {
  int i = blockIdx.x * blockDim.x + threadIdx.x;
  if (i >= rows) return;
  const unsigned short* r = c + (long)i * ld;
  float s = 0.f;
  for (int k = 0; k < D; ++k) { float v = bf2f(r[k]); s += v * v; }
  na[i] = s;
}

__global__ void norm4_kernel(const float* __restrict__ deg,
                             float* __restrict__ n4, int n) {
  int i = blockIdx.x * blockDim.x + threadIdx.x;
  if (i >= n) return;
  float d = deg[i] + 1e-6f;   // norm_k = (deg+1e-6)^(-K/2), K=4 -> d^-2
  float inv = 1.f / d;
  n4[i] = inv * inv;
}

__global__ void ghcombine_kernel(const unsigned short* __restrict__ gate,
                                 const unsigned short* __restrict__ fh,
                                 const unsigned short* __restrict__ hh,
                                 unsigned short* __restrict__ y,
                                 long ld, int rows, int width) {
  long i = (long)blockIdx.x * blockDim.x + threadIdx.x;
  long total = (long)rows * width;
  if (i >= total) return;
  long r = i / width; int c = (int)(i % width);
  long idx = r * ld + c;
  float g = bf2f(gate[idx]);
  float v = g * bf2f(fh[idx]) + (1.f - g) * bf2f(hh[idx]);
  y[idx] = f2bf(v);
}

__global__ void copycols_kernel(const float* __restrict__ src, int srcld,
                                unsigned short* __restrict__ dst, long dstld,
                                int coloff, int rows, int cols) {
  long i = (long)blockIdx.x * blockDim.x + threadIdx.x;
  long total = (long)rows * cols;
  if (i >= total) return;
  long r = i / cols; int c = (int)(i % cols);
  dst[r * dstld + coloff + c] = f2bf(src[(long)r * srcld + c]);
}

__global__ void output_kernel(const float* __restrict__ X,
                              const float* __restrict__ oid,
                              const float* __restrict__ chg,
                              const float* __restrict__ corr,
                              const float* __restrict__ msk,
                              float* __restrict__ out, int bn) {
  int i = blockIdx.x * blockDim.x + threadIdx.x;
  if (i >= bn) return;
  float m = msk[i];
  float* o = out + (long)i * 12;
#pragma unroll
  for (int j = 0; j < 8; ++j) o[j] = oid[(long)i * 8 + j] * m;
#pragma unroll
  for (int j = 0; j < 3; ++j) o[8 + j] = (X[(long)i * FFP + 2 + j] + corr[(long)i * 3 + j]) * m;
  o[11] = chg[i] * m;
}

// ---------------- generic WMMA GEMM ----------------
// C = act(A[M,K] @ W[K,N] + bias) * rowscale; bf16 in, f32 accum, bf16 out.
// Macro tile 32(M) x 128(N), 8 waves as 2x4, each wave owns two 16x16 tiles.
// act: 0=none, 1=selu, 2=sigmoid.  Batched via grid.z with element strides.
__global__ __launch_bounds__(256) void wmma_gemm_kernel(
    const unsigned short* __restrict__ A, long lda, long sA,
    const unsigned short* __restrict__ W, long ldw, long sW,
    const float* __restrict__ bias,
    unsigned short* __restrict__ C, long ldc, long sC,
    float* __restrict__ C2, long ldc2, long sC2,
    const float* __restrict__ rowscale, long sRS,
    int M, int Kd, int Nd, int act) {
  int bz = blockIdx.z;
  A += (long)bz * sA;  W += (long)bz * sW;  C += (long)bz * sC;
  if (C2)       C2       += (long)bz * sC2;
  if (rowscale) rowscale += (long)bz * sRS;

  __shared__ unsigned short As[32][40];    // 32(M) x 32(K)
  __shared__ unsigned short Wt[128][40];   // transposed: [n][k]

  int tid = threadIdx.x, lane = tid & 31, wave = tid >> 5;
  int wr = wave >> 2, wc = wave & 3;
  int m0 = blockIdx.y * 32, n0 = blockIdx.x * 128;
  int half = lane >> 4, l16 = lane & 15;

  f32x8 acc0 = {}, acc1 = {};
  for (int k0 = 0; k0 < Kd; k0 += 32) {
    __syncthreads();
    // ---- A tile ----
    if (m0 + 32 <= M && k0 + 32 <= Kd) {
      if (tid < 128) {
        int r = tid >> 2, c = (tid & 3) * 8;
        const unsigned short* gp = A + (long)(m0 + r) * lda + k0 + c;
        *(uint4*)&As[r][c] = *(const uint4*)gp;                 // b128 -> b128
        if (k0 + 64 <= Kd) __builtin_prefetch(gp + 32, 0, 3);   // global_prefetch
      }
    } else {
      for (int i = tid; i < 32 * 32; i += 256) {
        int r = i >> 5, c = i & 31, gm = m0 + r, gk = k0 + c;
        As[r][c] = (gm < M && gk < Kd) ? A[(long)gm * lda + gk] : (unsigned short)0;
      }
    }
    // ---- W tile (stored transposed) ----
    if (k0 + 32 <= Kd && n0 + 128 <= Nd) {
#pragma unroll
      for (int g = 0; g < 2; ++g) {
        int idx = tid + 256 * g;
        int k = idx >> 4, gn = (idx & 15) * 8;
        const unsigned short* gp = W + (long)(k0 + k) * ldw + n0 + gn;
        uint4 v = *(const uint4*)gp;                            // b128 load
        if (k0 + 64 <= Kd) __builtin_prefetch(gp + 32 * ldw, 0, 3);
        Wt[gn + 0][k] = (unsigned short)(v.x);
        Wt[gn + 1][k] = (unsigned short)(v.x >> 16);
        Wt[gn + 2][k] = (unsigned short)(v.y);
        Wt[gn + 3][k] = (unsigned short)(v.y >> 16);
        Wt[gn + 4][k] = (unsigned short)(v.z);
        Wt[gn + 5][k] = (unsigned short)(v.z >> 16);
        Wt[gn + 6][k] = (unsigned short)(v.w);
        Wt[gn + 7][k] = (unsigned short)(v.w >> 16);
      }
    } else {
      for (int i = tid; i < 32 * 128; i += 256) {
        int k = i >> 7, n = i & 127;
        int gk = k0 + k, gn = n0 + n;
        Wt[n][k] = (gk < Kd && gn < Nd) ? W[(long)gk * ldw + gn] : (unsigned short)0;
      }
    }
    __syncthreads();
    // ---- fragments: two ds_load_b128 each ----
    const unsigned short* ar = &As[wr * 16 + l16][0];
    bf16x16 af = frag2(ar + 8 * half, ar + 16 + 8 * half);
    const unsigned short* b0 = &Wt[wc * 32 + l16][8 * half];
    const unsigned short* b1 = &Wt[wc * 32 + 16 + l16][8 * half];
    bf16x16 bva = frag2(b0, b0 + 16);
    bf16x16 bvb = frag2(b1, b1 + 16);
    acc0 = __builtin_amdgcn_wmma_f32_16x16x32_bf16(false, af, false, bva,
                                                   (short)0, acc0, false, false);
    acc1 = __builtin_amdgcn_wmma_f32_16x16x32_bf16(false, af, false, bvb,
                                                   (short)0, acc1, false, false);
  }
  // ---- epilogue: lane%16 = N col, VGPR i -> M row = i + 8*half ----
  int colA = n0 + wc * 32 + l16;
  auto store_tile = [&](f32x8& acc, int col) {
    if (col >= Nd) return;
    float bb = bias ? bias[col] : 0.f;
#pragma unroll
    for (int i = 0; i < 8; ++i) {
      int row = m0 + wr * 16 + i + 8 * half;
      if (row < M) {
        float v = acc[i] + bb;
        if (act == 1)      v = selu_f(v);
        else if (act == 2) v = 1.f / (1.f + __expf(-v));
        if (rowscale)      v *= rowscale[row];
        C[(long)row * ldc + col] = f2bf(v);
        if (C2) C2[(long)row * ldc2 + col] = v;
      }
    }
  };
  store_tile(acc0, colA);
  store_tile(acc1, colA + 16);
}

// ---------------- WMMA adjacency kernel ----------------
// G = coords @ coordsT tile; adj4 = (exp(-sqrt(clip(na_r - 2G + na_c))) * msk_r)^4.
// Macro tile 32(rows) x 128(cols); always on fast path (4096/256 dims).
// Row-degree sums reduced via LDS ds_add_f32, then one global atomic per row.
__global__ __launch_bounds__(256) void adj_wmma_kernel(
    const unsigned short* __restrict__ coords, long ldc_, long sCo,
    const float* __restrict__ na, const float* __restrict__ msk,
    unsigned short* __restrict__ adj4, float* __restrict__ deg,
    int Nn, int Dd) {
  int bz = blockIdx.z;
  coords += (long)bz * sCo;
  const float* nab  = na  + (long)bz * Nn;
  const float* mskb = msk + (long)bz * Nn;
  unsigned short* adjb = adj4 + (long)bz * (long)Nn * Nn;
  float* degb = deg + (long)bz * Nn;

  __shared__ unsigned short As[32][40];    // row tile [m][k]
  __shared__ unsigned short Bs[128][40];   // col tile [n][k] (natural layout = B^T)
  __shared__ float degs[32];

  int tid = threadIdx.x, lane = tid & 31, wave = tid >> 5;
  int wr = wave >> 2, wc = wave & 3;
  int m0 = blockIdx.y * 32, n0 = blockIdx.x * 128;
  int half = lane >> 4, l16 = lane & 15;
  if (tid < 32) degs[tid] = 0.f;

  f32x8 acc0 = {}, acc1 = {};
  for (int k0 = 0; k0 < Dd; k0 += 32) {
    __syncthreads();
    if (tid < 128) {                       // A tile: 32x32
      int r = tid >> 2, c = (tid & 3) * 8;
      *(uint4*)&As[r][c] = *(const uint4*)(coords + (long)(m0 + r) * ldc_ + k0 + c);
    }
#pragma unroll
    for (int g = 0; g < 2; ++g) {          // B tile: 128 rows x 32 k
      int idx = tid + 256 * g;
      int r = idx >> 2, c = (idx & 3) * 8;
      *(uint4*)&Bs[r][c] = *(const uint4*)(coords + (long)(n0 + r) * ldc_ + k0 + c);
    }
    __syncthreads();
    const unsigned short* ar = &As[wr * 16 + l16][0];
    bf16x16 af = frag2(ar + 8 * half, ar + 16 + 8 * half);
    const unsigned short* b0 = &Bs[wc * 32 + l16][8 * half];
    const unsigned short* b1 = &Bs[wc * 32 + 16 + l16][8 * half];
    bf16x16 bva = frag2(b0, b0 + 16);
    bf16x16 bvb = frag2(b1, b1 + 16);
    acc0 = __builtin_amdgcn_wmma_f32_16x16x32_bf16(false, af, false, bva,
                                                   (short)0, acc0, false, false);
    acc1 = __builtin_amdgcn_wmma_f32_16x16x32_bf16(false, af, false, bvb,
                                                   (short)0, acc1, false, false);
  }
  int colA = n0 + wc * 32 + l16;
  auto emit = [&](f32x8& acc, int col) {
    if (col >= Nn) return;
    float nac = nab[col];
#pragma unroll
    for (int i = 0; i < 8; ++i) {
      int row = m0 + wr * 16 + i + 8 * half;
      if (row < Nn) {
        float dsq = nab[row] - 2.f * acc[i] + nac;
        dsq = fminf(fmaxf(dsq, 1e-12f), 1e12f);
        float e = __expf(-sqrtf(dsq)) * mskb[row];
        float e2 = e * e;
        adjb[(long)row * Nn + col] = f2bf(e2 * e2);      // adj^K, K=4
        atomicAdd(&degs[wr * 16 + i + 8 * half], e);     // ds_add_f32
      }
    }
  };
  emit(acc0, colA);
  emit(acc1, colA + 16);
  __syncthreads();
  if (tid < 32) {
    int row = m0 + tid;
    if (row < Nn) atomicAdd(&degb[row], degs[tid]);      // global_atomic_add_f32
  }
}

// ---------------- host orchestration ----------------
extern "C" void kernel_launch(void* const* d_in, const int* in_sizes, int n_in,
                              void* d_out, int out_size, void* d_ws, size_t ws_size,
                              hipStream_t stream) {
  (void)in_sizes; (void)n_in; (void)out_size; (void)ws_size;
  const float* X = (const float*)d_in[0];
  const float* Wf[19]; const float* Bf[19];
  for (int i = 0; i < 19; ++i) { Wf[i] = (const float*)d_in[1 + 2 * i]; Bf[i] = (const float*)d_in[2 + 2 * i]; }
  const float* g1_wt = (const float*)d_in[39];
  const float* g1_wh = (const float*)d_in[40];
  const float* g1_th = (const float*)d_in[41];
  const float* g1_bt = (const float*)d_in[42];
  const float* g2_wt = (const float*)d_in[43];
  const float* g2_wh = (const float*)d_in[44];
  const float* g2_th = (const float*)d_in[45];
  const float* g2_bt = (const float*)d_in[46];

  static const int WDI[19] = {27,256,256,256, 27,256,256, 512,512,256, 256,256, 27,256,256, 520,512,256, 256};
  static const int WDO[19] = {256,256,256,256, 256,256,512, 512,256,256, 8,1, 256,256,512, 512,256,256, 3};

  char* base = (char*)d_ws; size_t off = 0;
  auto alloc = [&](size_t bytes) -> void* {
    void* p = base + off; off = (off + bytes + 255) & ~(size_t)255; return p;
  };

  unsigned short* Wd[19];
  for (int i = 0; i < 19; ++i) Wd[i] = (unsigned short*)alloc((size_t)WDI[i] * WDO[i] * 2);
  unsigned short* G1wt = (unsigned short*)alloc((size_t)H2P * H2P * 2);
  unsigned short* G1wh = (unsigned short*)alloc((size_t)H2P * H2P * 2);
  unsigned short* G1th = (unsigned short*)alloc((size_t)H2P * H2P * 2);
  unsigned short* G2wt = (unsigned short*)alloc((size_t)H2CP * H2CP * 2);
  unsigned short* G2wh = (unsigned short*)alloc((size_t)H2CP * H2CP * 2);
  unsigned short* G2th = (unsigned short*)alloc((size_t)H2CP * H2CP * 2);

  unsigned short* enc = (unsigned short*)alloc((size_t)BNP * 32 * 2);
  float* msk   = (float*)alloc((size_t)BNP * 4);
  float* na    = (float*)alloc((size_t)BNP * 4);
  float* deg   = (float*)alloc((size_t)BNP * 4);
  float* nrm4  = (float*)alloc((size_t)BNP * 4);
  float* oidf  = (float*)alloc((size_t)BNP * 8 * 4);
  float* chgf  = (float*)alloc((size_t)BNP * 4);
  float* corrf = (float*)alloc((size_t)BNP * 3 * 4);
  unsigned short* adj4 = (unsigned short*)alloc((size_t)BB * NNP * NNP * 2);  // 64 MiB, L2-resident
  unsigned short* Bb[5];
  for (int i = 0; i < 5; ++i) Bb[i] = (unsigned short*)alloc((size_t)BNP * LDP * 2);

  auto cvt = [&](const float* s, unsigned short* d, long n) {
    cvt_f32_bf16_kernel<<<dim3((unsigned)((n + 255) / 256)), dim3(256), 0, stream>>>(s, d, n);
  };
  for (int i = 0; i < 19; ++i) cvt(Wf[i], Wd[i], (long)WDI[i] * WDO[i]);
  cvt(g1_wt, G1wt, (long)H2P * H2P);  cvt(g1_wh, G1wh, (long)H2P * H2P);  cvt(g1_th, G1th, (long)H2P * H2P);
  cvt(g2_wt, G2wt, (long)H2CP * H2CP); cvt(g2_wh, G2wh, (long)H2CP * H2CP); cvt(g2_th, G2th, (long)H2CP * H2CP);

  auto gemm = [&](const unsigned short* A, long lda, long sA,
                  const unsigned short* W, long ldw, long sW,
                  const float* bias,
                  unsigned short* C, long ldc, long sC,
                  float* C2, long ldc2, long sC2,
                  const float* rs, long sRS,
                  int M, int K, int N, int act, int batches) {
    dim3 g((unsigned)((N + 127) / 128), (unsigned)((M + 31) / 32), (unsigned)batches);
    wmma_gemm_kernel<<<g, 256, 0, stream>>>(A, lda, sA, W, ldw, sW, bias,
                                            C, ldc, sC, C2, ldc2, sC2, rs, sRS, M, K, N, act);
  };

  // ---- encode ----
  encode_kernel<<<(BNP + 255) / 256, 256, 0, stream>>>(X, enc, msk, BNP);

  // ---- phase A: coords ----
  gemm(enc, 32, 0, Wd[0], 256, 0, Bf[0], Bb[0], LDP, 0, 0, 0, 0, 0, 0, BNP, 27, 256, 1, 1);
  gemm(Bb[0], LDP, 0, Wd[1], 256, 0, Bf[1], Bb[1], LDP, 0, 0, 0, 0, 0, 0, BNP, 256, 256, 1, 1);
  gemm(Bb[1], LDP, 0, Wd[2], 256, 0, Bf[2], Bb[0], LDP, 0, 0, 0, 0, 0, 0, BNP, 256, 256, 1, 1);
  gemm(Bb[0], LDP, 0, Wd[3], 256, 0, Bf[3], Bb[1], LDP, 0, 0, 0, 0, 0, 0, BNP, 256, 256, 0, 1);  // coords->B1

  // ---- phase B: adjacency^4 (bf16) + degree ----
  rowsq_kernel<<<(BNP + 255) / 256, 256, 0, stream>>>(Bb[1], LDP, na, BNP, 256);
  zero_f32_kernel<<<(BNP + 255) / 256, 256, 0, stream>>>(deg, BNP);
  adj_wmma_kernel<<<dim3(NNP / 128, NNP / 32, BB), 256, 0, stream>>>(
      Bb[1], LDP, (long)NNP * LDP, na, msk, adj4, deg, NNP, 256);
  norm4_kernel<<<(BNP + 255) / 256, 256, 0, stream>>>(deg, nrm4, BNP);

  // ---- phase C: id branch + GHConv g1 ----
  gemm(enc, 32, 0, Wd[4], 256, 0, Bf[4], Bb[0], LDP, 0, 0, 0, 0, 0, 0, BNP, 27, 256, 1, 1);
  gemm(Bb[0], LDP, 0, Wd[5], 256, 0, Bf[5], Bb[2], LDP, 0, 0, 0, 0, 0, 0, BNP, 256, 256, 1, 1);
  gemm(Bb[2], LDP, 0, Wd[6], 512, 0, Bf[6], Bb[0], LDP, 0, 0, 0, 0, 0, 0, BNP, 256, 512, 1, 1);  // x->B0
  gemm(Bb[0], LDP, 0, G1th, 512, 0, nullptr, Bb[2], LDP, 0, 0, 0, 0, nrm4, 0, BNP, 512, 512, 0, 1);   // th
  gemm(adj4, NNP, (long)NNP * NNP, Bb[2], LDP, (long)NNP * LDP, nullptr,
       Bb[3], LDP, (long)NNP * LDP, 0, 0, 0, nrm4, NNP, NNP, NNP, 512, 0, BB);                        // f_hom
  gemm(Bb[0], LDP, 0, G1wt, 512, 0, g1_bt, Bb[4], LDP, 0, 0, 0, 0, 0, 0, BNP, 512, 512, 2, 1);        // gate
  gemm(Bb[0], LDP, 0, G1wh, 512, 0, nullptr, Bb[2], LDP, 0, 0, 0, 0, 0, 0, BNP, 512, 512, 0, 1);      // x@wh
  ghcombine_kernel<<<(unsigned)(((long)BNP * 512 + 255) / 256), 256, 0, stream>>>(
      Bb[4], Bb[3], Bb[2], Bb[0], LDP, BNP, 512);
  gemm(Bb[0], LDP, 0, Wd[7], 512, 0, Bf[7], Bb[1], LDP, 0, 0, 0, 0, 0, 0, BNP, 512, 512, 1, 1);
  gemm(Bb[1], LDP, 0, Wd[8], 256, 0, Bf[8], Bb[2], LDP, 0, 0, 0, 0, 0, 0, BNP, 512, 256, 1, 1);
  gemm(Bb[2], LDP, 0, Wd[9], 256, 0, Bf[9], Bb[1], LDP, 0, 0, 0, 0, 0, 0, BNP, 256, 256, 1, 1);
  gemm(Bb[1], LDP, 0, Wd[10], 8, 0, Bf[10], Bb[3], LDP, 0, oidf, 8, 0, 0, 0, BNP, 256, 8, 0, 1);      // oid
  gemm(Bb[1], LDP, 0, Wd[11], 1, 0, Bf[11], Bb[2], LDP, 0, chgf, 1, 0, 0, 0, BNP, 256, 1, 0, 1);      // chg

  // ---- phase D: momentum branch + GHConv g2 ----
  gemm(enc, 32, 0, Wd[12], 256, 0, Bf[12], Bb[0], LDP, 0, 0, 0, 0, 0, 0, BNP, 27, 256, 1, 1);
  gemm(Bb[0], LDP, 0, Wd[13], 256, 0, Bf[13], Bb[2], LDP, 0, 0, 0, 0, 0, 0, BNP, 256, 256, 1, 1);
  gemm(Bb[2], LDP, 0, Wd[14], 512, 0, Bf[14], Bb[3], LDP, 0, 0, 0, 0, 0, 0, BNP, 256, 512, 1, 1);     // xm->B3
  copycols_kernel<<<(unsigned)(((long)BNP * 8 + 255) / 256), 256, 0, stream>>>(
      oidf, 8, Bb[3], LDP, 512, BNP, 8);                                                              // concat -> 520
  gemm(Bb[3], LDP, 0, G2th, 520, 0, nullptr, Bb[0], LDP, 0, 0, 0, 0, nrm4, 0, BNP, 520, 520, 0, 1);   // th2
  gemm(adj4, NNP, (long)NNP * NNP, Bb[0], LDP, (long)NNP * LDP, nullptr,
       Bb[1], LDP, (long)NNP * LDP, 0, 0, 0, nrm4, NNP, NNP, NNP, 520, 0, BB);                        // f_hom2
  gemm(Bb[3], LDP, 0, G2wt, 520, 0, g2_bt, Bb[2], LDP, 0, 0, 0, 0, 0, 0, BNP, 520, 520, 2, 1);        // gate2
  gemm(Bb[3], LDP, 0, G2wh, 520, 0, nullptr, Bb[4], LDP, 0, 0, 0, 0, 0, 0, BNP, 520, 520, 0, 1);      // xm@wh
  ghcombine_kernel<<<(unsigned)(((long)BNP * 520 + 255) / 256), 256, 0, stream>>>(
      Bb[2], Bb[1], Bb[4], Bb[0], LDP, BNP, 520);
  gemm(Bb[0], LDP, 0, Wd[15], 512, 0, Bf[15], Bb[1], LDP, 0, 0, 0, 0, 0, 0, BNP, 520, 512, 1, 1);
  gemm(Bb[1], LDP, 0, Wd[16], 256, 0, Bf[16], Bb[0], LDP, 0, 0, 0, 0, 0, 0, BNP, 512, 256, 1, 1);
  gemm(Bb[0], LDP, 0, Wd[17], 256, 0, Bf[17], Bb[1], LDP, 0, 0, 0, 0, 0, 0, BNP, 256, 256, 1, 1);
  gemm(Bb[1], LDP, 0, Wd[18], 3, 0, Bf[18], Bb[0], LDP, 0, corrf, 3, 0, 0, 0, BNP, 256, 3, 0, 1);     // corr

  // ---- output ----
  output_kernel<<<(BNP + 255) / 256, 256, 0, stream>>>(X, oidf, chgf, corrf, msk, (float*)d_out, BNP);
}
